// HeteroGATv2Encoder_89146341195957
// MI455X (gfx1250) — matrix operand order repaired and checked by reading, not compiled, not run on hardware
//
#include <hip/hip_runtime.h>
#include <cmath>

#define NNODE 20000
#define NEDGE 100000
#define DH    512
#define H01   8
#define C01   64
#define DOUT  256

typedef __attribute__((ext_vector_type(16))) __bf16 v16bf;
typedef __attribute__((ext_vector_type(8)))  float  v8f;
typedef __attribute__((ext_vector_type(4)))  float  vf4;

union Frag32 { v16bf v; vf4 f[2]; };
static_assert(sizeof(v16bf) == 32, "bf16x16 fragment must be 32B");

__device__ __forceinline__ unsigned short f2bf(float f) {
  unsigned u = __float_as_uint(f);
  u += 0x7FFFu + ((u >> 16) & 1u);          // round-to-nearest-even
  return (unsigned short)(u >> 16);
}
__device__ __forceinline__ float eluf(float v)  { return v > 0.f ? v : (__expf(v) - 1.f); }
__device__ __forceinline__ float lrelu(float v) { return v > 0.f ? v : 0.2f * v; }

// float atomic max via signed-max / unsigned-min bit trick (init = -inf)
__device__ __forceinline__ void atomicMaxF(float* addr, float val) {
  if (val >= 0.f) atomicMax((int*)addr, __float_as_int(val));
  else            atomicMin((unsigned int*)addr, __float_as_uint(val));
}

// ---------------- WMMA GEMM: C[M x NOUT] = A[M x 512](bf16) * B(bf16, pre-transposed) + bias ----
// The 16x512 A tile is staged once per block into LDS with CDNA5 async-to-LDS copies
// (ASYNCcnt), padded to 520 elem/row so the 16-lane b128 fragment reads are bank-conflict-free.
// One wave computes a 16x64 tile: A-fragment reused across 4 v_wmma_f32_16x16x32_bf16.
template<int NOUT, bool ELU>
__global__ void gemm_bf16_k(const unsigned short* __restrict__ A,
                            const unsigned short* __restrict__ Bt,   // [NOUT x 512] = B^T
                            const float* __restrict__ bias,
                            float* __restrict__ C) {
  constexpr int K = DH;
  constexpr int LROW = 520;                   // padded row stride (bf16 elems): 1040B = 4 banks skew
  constexpr int NT = (NOUT / 64) * 32;        // threads per block
  __shared__ unsigned short tileA[16 * LROW];

  const int tid  = threadIdx.x;
  const int lane = tid & 31;
  const int wv   = tid >> 5;
  const int m0   = blockIdx.x * 16;
  const int n0   = wv * 64;
  const bool hi  = lane >= 16;
  const int l15  = lane & 15;

  // ---- cooperative async copy of A tile (16 rows x 1KB) into LDS ----
  {
    unsigned ldsbase = (unsigned)(size_t)&tileA[0];
    const unsigned short* gbase = A + (size_t)m0 * K;
#pragma unroll
    for (int q = tid; q < 16 * 64; q += NT) {      // 1024 x 16B chunks
      int row = q >> 6, col = q & 63;
      unsigned loff = ldsbase + (unsigned)(row * (LROW * 2) + col * 16);
      unsigned long long ga =
          (unsigned long long)(const void*)(gbase + row * K + col * 8);
      asm volatile("global_load_async_to_lds_b128 %0, %1, off"
                   :: "v"(loff), "v"(ga) : "memory");
    }
    asm volatile("s_wait_asynccnt 0x0" ::: "memory");
    __syncthreads();
  }

  // ISA 7.12.2 16-bit A 16x32: lanes 0-15 row M=lane K={0..7,16..23}; lanes 16-31 K={8..15,24..31}
  const unsigned short* arow = &tileA[l15 * LROW + (hi ? 8 : 0)];

  v8f zero = {};
  v8f acc[4];
#pragma unroll
  for (int t = 0; t < 4; ++t) acc[t] = zero;

  for (int k0 = 0; k0 < K; k0 += 32) {
    Frag32 a;
    a.f[0] = *(const vf4*)(arow + k0);        // ds_load_b128, conflict-free
    a.f[1] = *(const vf4*)(arow + k0 + 16);
    const int kb = k0 + (hi ? 8 : 0);         // B: lanes 0-15 K={0..7,16..23}, lanes 16-31 K={8..15,24..31}
#pragma unroll
    for (int t = 0; t < 4; ++t) {
      const unsigned short* bptr = Bt + (size_t)(n0 + t * 16 + l15) * K + kb;
      Frag32 b;
      b.f[0] = *(const vf4*)(bptr);
      b.f[1] = *(const vf4*)(bptr + 16);
      acc[t] = __builtin_amdgcn_wmma_f32_16x16x32_bf16(false, a.v, false, b.v,
                                                       (short)0, acc[t], false, false);
    }
  }

  // C/D layout: lane<16 -> (M=r, N=lane); lane>=16 -> (M=r+8, N=lane-16)
  const int mbase = m0 + (hi ? 8 : 0);
#pragma unroll
  for (int t = 0; t < 4; ++t) {
    const int ncol = n0 + t * 16 + l15;
    const float bv = bias[ncol];
#pragma unroll
    for (int r = 0; r < 8; ++r) {
      float v = acc[t][r] + bv;
      if (ELU) v = eluf(v);
      C[(size_t)(mbase + r) * NOUT + ncol] = v;
    }
  }
}

// ---------------- edge kernels (segment softmax in 3 atomic passes) ----------------
template<int H, int C>
__global__ void edge_score_k(const float* __restrict__ xl, const float* __restrict__ xr,
                             const int* __restrict__ src, const int* __restrict__ dst,
                             const float* __restrict__ att,
                             float* __restrict__ score, float* __restrict__ mmax) {
  int t = blockIdx.x * blockDim.x + threadIdx.x;
  if (t >= NEDGE * H) return;
  int e = t / H, h = t - e * H;
  int s = src[e], d = dst[e];
  const float* pl = xl + (size_t)s * (H * C) + h * C;
  const float* pr = xr + (size_t)d * (H * C) + h * C;
  const float* pa = att + h * C;
  float acc = 0.f;
#pragma unroll 4
  for (int c = 0; c < C; c += 4) {
    vf4 a = *(const vf4*)(pl + c);
    vf4 b = *(const vf4*)(pr + c);
    vf4 w = *(const vf4*)(pa + c);
    acc += w.x * lrelu(a.x + b.x);
    acc += w.y * lrelu(a.y + b.y);
    acc += w.z * lrelu(a.z + b.z);
    acc += w.w * lrelu(a.w + b.w);
  }
  score[t] = acc;
  atomicMaxF(&mmax[(size_t)d * H + h], acc);
}

__global__ void edge_exp_k(float* __restrict__ score, const float* __restrict__ mmax,
                           float* __restrict__ zsum, const int* __restrict__ dst, int H) {
  int t = blockIdx.x * blockDim.x + threadIdx.x;
  if (t >= NEDGE * H) return;
  int e = t / H, h = t - e * H;
  int d = dst[e];
  float w = __expf(score[t] - mmax[(size_t)d * H + h]);
  score[t] = w;                       // reuse buffer as un-normalized weight
  atomicAdd(&zsum[(size_t)d * H + h], w);
}

template<int H, int C>
__global__ void edge_scatter_k(const float* __restrict__ xl, const int* __restrict__ src,
                               const int* __restrict__ dst, const float* __restrict__ wgt,
                               const float* __restrict__ zsum, float* __restrict__ out) {
  constexpr int D = H * C;
  constexpr int Q = D / 4;            // float4 chunks per edge
  int t = blockIdx.x * blockDim.x + threadIdx.x;
  if (t >= NEDGE * Q) return;
  int e = t / Q, q = t - e * Q;
  int c = q * 4, h = c / C;
  int s = src[e], d = dst[e];
  float alpha = wgt[(size_t)e * H + h] / zsum[(size_t)d * H + h];
  vf4 xv = *(const vf4*)(xl + (size_t)s * D + c);
  float* po = out + (size_t)d * D + c;
  atomicAdd(po + 0, alpha * xv.x);
  atomicAdd(po + 1, alpha * xv.y);
  atomicAdd(po + 2, alpha * xv.z);
  atomicAdd(po + 3, alpha * xv.w);
}

// ---------------- elementwise / prep kernels ----------------
__global__ void to_bf16_k(const float* __restrict__ in, unsigned short* __restrict__ out, int n) {
  int i = blockIdx.x * blockDim.x + threadIdx.x;
  if (i < n) out[i] = f2bf(in[i]);
}
__global__ void transpose_bf16_k(const float* __restrict__ W, unsigned short* __restrict__ Wt,
                                 int K, int NO) {   // W[K x NO] -> Wt[NO x K] bf16
  int idx = blockIdx.x * blockDim.x + threadIdx.x;
  if (idx >= K * NO) return;
  int n = idx / K, k = idx - n * K;
  Wt[idx] = f2bf(W[(size_t)k * NO + n]);
}
__global__ void fill_k(float* __restrict__ p, float v, int n) {
  int i = blockIdx.x * blockDim.x + threadIdx.x;
  if (i < n) p[i] = v;
}
__global__ void init_bias3_k(float* __restrict__ out, const float* __restrict__ b, int D, int total) {
  int i = blockIdx.x * blockDim.x + threadIdx.x;
  if (i >= total) return;
  int c = i % D;
  out[i] = b[c] + b[D + c] + b[2 * D + c];   // sum of the 3 relation output-biases
}
__global__ void elu_add_k(const float* __restrict__ o, float* __restrict__ h, int n) {
  int i = blockIdx.x * blockDim.x + threadIdx.x;
  if (i < n) h[i] = eluf(o[i]) + h[i];
}

template<bool ELU_IN>
__global__ void ln_k(const float* __restrict__ in, float* __restrict__ out,
                     const float* __restrict__ g, const float* __restrict__ b, int D) {
  __shared__ float red[128];
  int row = blockIdx.x;
  const float* p = in + (size_t)row * D;
  float s = 0.f, s2 = 0.f;
  for (int c = threadIdx.x; c < D; c += 128) {
    float v = p[c];
    if (ELU_IN) v = eluf(v);
    s += v; s2 += v * v;
  }
  red[threadIdx.x] = s; __syncthreads();
  for (int o = 64; o > 0; o >>= 1) { if (threadIdx.x < o) red[threadIdx.x] += red[threadIdx.x + o]; __syncthreads(); }
  float mean = red[0] / D;
  __syncthreads();
  red[threadIdx.x] = s2; __syncthreads();
  for (int o = 64; o > 0; o >>= 1) { if (threadIdx.x < o) red[threadIdx.x] += red[threadIdx.x + o]; __syncthreads(); }
  float var = red[0] / D - mean * mean;
  float inv = rsqrtf(var + 1e-5f);
  for (int c = threadIdx.x; c < D; c += 128) {
    float v = p[c];
    if (ELU_IN) v = eluf(v);
    out[(size_t)row * D + c] = (v - mean) * inv * g[c] + b[c];
  }
}

// ---------------- host orchestration ----------------
extern "C" void kernel_launch(void* const* d_in, const int* in_sizes, int n_in,
                              void* d_out, int out_size, void* d_ws, size_t ws_size,
                              hipStream_t stream) {
  (void)in_sizes; (void)n_in; (void)out_size; (void)ws_size;
  const float* x        = (const float*)d_in[0];
  const int*   eidx     = (const int*)  d_in[1];
  const float* proj_w   = (const float*)d_in[2];
  const float* proj_b   = (const float*)d_in[3];
  const float* ll_w     = (const float*)d_in[4];
  const float* ll_b     = (const float*)d_in[5];
  const float* lr_w     = (const float*)d_in[6];
  const float* lr_b     = (const float*)d_in[7];
  const float* l01_att  = (const float*)d_in[8];
  const float* l01_bias = (const float*)d_in[9];
  const float* l2l_w    = (const float*)d_in[10];
  const float* l2l_b    = (const float*)d_in[11];
  const float* l2r_w    = (const float*)d_in[12];
  const float* l2r_b    = (const float*)d_in[13];
  const float* l2_att   = (const float*)d_in[14];
  const float* l2_bias  = (const float*)d_in[15];
  const float* ln01_g   = (const float*)d_in[16];
  const float* ln01_b   = (const float*)d_in[17];
  const float* ln2_g    = (const float*)d_in[18];
  const float* ln2_b    = (const float*)d_in[19];

  char* base = (char*)d_ws;
  size_t off = 0;
  auto walloc = [&](size_t bytes) -> void* {
    void* r = base + off; off = (off + bytes + 255) & ~(size_t)255; return r;
  };
  float*          h     = (float*)walloc((size_t)NNODE * DH * 4);
  float*          outac = (float*)walloc((size_t)NNODE * DH * 4);
  float*          xl    = (float*)walloc((size_t)NNODE * DH * 4);
  float*          xr    = (float*)walloc((size_t)NNODE * DH * 4);
  unsigned short* hbf   = (unsigned short*)walloc((size_t)NNODE * DH * 2);
  unsigned short* wt    = (unsigned short*)walloc((size_t)DH * DH * 2);
  float*          score = (float*)walloc((size_t)NEDGE * H01 * 4);
  float*          mbuf  = (float*)walloc((size_t)NNODE * H01 * 4);
  float*          zbuf  = (float*)walloc((size_t)NNODE * H01 * 4);

  const int TB = 256;
  auto cdiv = [](long long a, long long b) { return (unsigned)((a + b - 1) / b); };

  // ---- input projection: h = elu(x @ proj_w + proj_b)
  to_bf16_k<<<cdiv((long long)NNODE * DH, TB), TB, 0, stream>>>(x, hbf, NNODE * DH);
  transpose_bf16_k<<<cdiv(DH * DH, TB), TB, 0, stream>>>(proj_w, wt, DH, DH);
  gemm_bf16_k<DH, true><<<NNODE / 16, (DH / 64) * 32, 0, stream>>>(hbf, wt, proj_b, h);

  // ---- layers 0 and 1 (8 heads x 64 ch, sum over 3 relations, ELU + residual + LN)
  for (int li = 0; li < 2; ++li) {
    to_bf16_k<<<cdiv((long long)NNODE * DH, TB), TB, 0, stream>>>(h, hbf, NNODE * DH);
    init_bias3_k<<<cdiv((long long)NNODE * DH, TB), TB, 0, stream>>>(
        outac, l01_bias + (size_t)li * 3 * DH, DH, NNODE * DH);
    for (int et = 0; et < 3; ++et) {
      const int* src = eidx + (size_t)et * 2 * NEDGE;
      const int* dst = src + NEDGE;
      const float* Wl = ll_w + (size_t)(li * 3 + et) * DH * DH;
      const float* Wr = lr_w + (size_t)(li * 3 + et) * DH * DH;
      const float* bl = ll_b + (size_t)(li * 3 + et) * DH;
      const float* br = lr_b + (size_t)(li * 3 + et) * DH;
      const float* at = l01_att + (size_t)(li * 3 + et) * H01 * C01;

      transpose_bf16_k<<<cdiv(DH * DH, TB), TB, 0, stream>>>(Wl, wt, DH, DH);
      gemm_bf16_k<DH, false><<<NNODE / 16, (DH / 64) * 32, 0, stream>>>(hbf, wt, bl, xl);
      transpose_bf16_k<<<cdiv(DH * DH, TB), TB, 0, stream>>>(Wr, wt, DH, DH);
      gemm_bf16_k<DH, false><<<NNODE / 16, (DH / 64) * 32, 0, stream>>>(hbf, wt, br, xr);

      fill_k<<<cdiv(NNODE * H01, TB), TB, 0, stream>>>(mbuf, -INFINITY, NNODE * H01);
      fill_k<<<cdiv(NNODE * H01, TB), TB, 0, stream>>>(zbuf, 0.f, NNODE * H01);
      edge_score_k<H01, C01><<<cdiv((long long)NEDGE * H01, TB), TB, 0, stream>>>(
          xl, xr, src, dst, at, score, mbuf);
      edge_exp_k<<<cdiv((long long)NEDGE * H01, TB), TB, 0, stream>>>(score, mbuf, zbuf, dst, H01);
      edge_scatter_k<H01, C01><<<cdiv((long long)NEDGE * (DH / 4), TB), TB, 0, stream>>>(
          xl, src, dst, score, zbuf, outac);
    }
    elu_add_k<<<cdiv((long long)NNODE * DH, TB), TB, 0, stream>>>(outac, h, NNODE * DH);
    ln_k<false><<<NNODE, 128, 0, stream>>>(h, h, ln01_g + (size_t)li * DH, ln01_b + (size_t)li * DH, DH);
  }

  // ---- layer 2 (1 head x 256 ch), ELU then LN -> d_out
  to_bf16_k<<<cdiv((long long)NNODE * DH, TB), TB, 0, stream>>>(h, hbf, NNODE * DH);
  init_bias3_k<<<cdiv((long long)NNODE * DOUT, TB), TB, 0, stream>>>(outac, l2_bias, DOUT, NNODE * DOUT);
  for (int et = 0; et < 3; ++et) {
    const int* src = eidx + (size_t)et * 2 * NEDGE;
    const int* dst = src + NEDGE;
    transpose_bf16_k<<<cdiv(DH * DOUT, TB), TB, 0, stream>>>(l2l_w + (size_t)et * DH * DOUT, wt, DH, DOUT);
    gemm_bf16_k<DOUT, false><<<NNODE / 16, (DOUT / 64) * 32, 0, stream>>>(hbf, wt, l2l_b + et * DOUT, xl);
    transpose_bf16_k<<<cdiv(DH * DOUT, TB), TB, 0, stream>>>(l2r_w + (size_t)et * DH * DOUT, wt, DH, DOUT);
    gemm_bf16_k<DOUT, false><<<NNODE / 16, (DOUT / 64) * 32, 0, stream>>>(hbf, wt, l2r_b + et * DOUT, xr);

    fill_k<<<cdiv(NNODE, TB), TB, 0, stream>>>(mbuf, -INFINITY, NNODE);
    fill_k<<<cdiv(NNODE, TB), TB, 0, stream>>>(zbuf, 0.f, NNODE);
    edge_score_k<1, DOUT><<<cdiv(NEDGE, TB), TB, 0, stream>>>(
        xl, xr, src, dst, l2_att + (size_t)et * DOUT, score, mbuf);
    edge_exp_k<<<cdiv(NEDGE, TB), TB, 0, stream>>>(score, mbuf, zbuf, dst, 1);
    edge_scatter_k<1, DOUT><<<cdiv((long long)NEDGE * (DOUT / 4), TB), TB, 0, stream>>>(
        xl, src, dst, score, zbuf, outac);
  }
  ln_k<true><<<NNODE, 128, 0, stream>>>(outac, (float*)d_out, ln2_g, ln2_b, DOUT);
}
